// TrackingNet_9758165697242
// MI455X (gfx1250) — compile-verified, matrix operand order checked by reading, not run on hardware
//
#include <hip/hip_runtime.h>
#include <hip/hip_bf16.h>

// DGCNN forward for MI455X (gfx1250, wave32).
// All dense contractions use v_wmma_f32_16x16x32_f16 (f16 inputs, f32 accum).

#define BB 8
#define NN 4096
#define KNN 20

typedef __attribute__((ext_vector_type(16))) _Float16 v16h;
typedef __attribute__((ext_vector_type(8)))  _Float16 v8h;
typedef __attribute__((ext_vector_type(8)))  float    v8f;

// Load a 16x16x32 WMMA A/B fragment from a row-major f16 LDS buffer.
// Layout (CDNA5 ISA 7.12.2, 16-bit A 16x32): lane l holds row (l&15),
// K = {8*kh..8*kh+7} and {16+8*kh..16+8*kh+7} where kh = l>>4.
// B uses the symmetric layout on a transposed (N x K row-major) buffer.
__device__ __forceinline__ v16h ld_frag(const _Float16* buf, int row0, int k0, int ld) {
  const int lane = threadIdx.x & 31;
  const int r  = row0 + (lane & 15);
  const int kh = lane >> 4;
  const _Float16* p = buf + r * ld + k0 + 8 * kh;
  v8h lo = *(const v8h*)(p);
  v8h hi = *(const v8h*)(p + 16);
  v16h out;
#pragma unroll
  for (int i = 0; i < 8; ++i) { out[i] = lo[i]; out[8 + i] = hi[i]; }
  return out;
}

__device__ __forceinline__ v8f wmma16(v16h a, v16h b, v8f c) {
  return __builtin_amdgcn_wmma_f32_16x16x32_f16(false, a, false, b, (short)0, c,
                                                false, false);
}

// ---------------------------------------------------------------- KNN #1 (2D)
__global__ __launch_bounds__(128) void knn1_kernel(const float* __restrict__ data,
                                                   int* __restrict__ idx1) {
  __shared__ float pts[NN * 2];
  const int b = blockIdx.x;
  const int i = blockIdx.y * blockDim.x + threadIdx.x;
  const float* src = data + (size_t)b * NN * 2;
  for (int t = threadIdx.x; t < NN * 2; t += blockDim.x) pts[t] = src[t];
  __syncthreads();
  const float xi = pts[2 * i], yi = pts[2 * i + 1];
  float bd[KNN]; int bi[KNN];
#pragma unroll
  for (int k = 0; k < KNN; ++k) { bd[k] = 3.4e38f; bi[k] = 0; }
  for (int j = 0; j < NN; ++j) {
    const float dx = pts[2 * j] - xi, dy = pts[2 * j + 1] - yi;
    const float d = dx * dx + dy * dy;
    if (d < bd[KNN - 1]) {
      int p = KNN - 1;
      while (p > 0 && bd[p - 1] > d) { bd[p] = bd[p - 1]; bi[p] = bi[p - 1]; --p; }
      bd[p] = d; bi[p] = j;
    }
  }
  int* o = idx1 + ((size_t)b * NN + i) * KNN;
  for (int k = 0; k < KNN; ++k) o[k] = bi[k];
}

// ------------------------------------------------------- EdgeConv #1 (4->64^3)
// 8 points per WG -> 160 edge rows; 3 GEMM layers fully in LDS.
__global__ __launch_bounds__(256) void edgeconv1_kernel(
    const float* __restrict__ data, const int* __restrict__ idx1,
    const float* __restrict__ w1, const float* __restrict__ b1,
    const float* __restrict__ w2, const float* __restrict__ b2,
    const float* __restrict__ w3, const float* __restrict__ b3,
    float* __restrict__ x1f, _Float16* __restrict__ x1h) {
  __shared__ alignas(16) _Float16 feat[160 * 32];   // layer-1 input, K padded to 32
  __shared__ alignas(16) _Float16 bufA[160 * 64];
  __shared__ alignas(16) _Float16 bufB[160 * 64];
  __shared__ alignas(16) _Float16 wbuf[64 * 64];    // transposed weights (N x K)
  const int tid = threadIdx.x, wid = tid >> 5, lane = tid & 31;
  const int blk = blockIdx.x;
  const int b = blk >> 9;               // 512 blocks per batch
  const int pbase = (blk & 511) * 8;

  for (int t = tid; t < 160 * 32; t += 256) feat[t] = (_Float16)0.f;
  for (int t = tid; t < 64 * 64; t += 256) wbuf[t] = (_Float16)0.f;
  __syncthreads();
  if (tid < 160) {
    const int p = tid / KNN, n = pbase + p;
    const int j = idx1[((size_t)b * NN + n) * KNN + (tid % KNN)];
    const float* pi = data + ((size_t)b * NN + n) * 2;
    const float* pj = data + ((size_t)b * NN + j) * 2;
    _Float16* fr = feat + tid * 32;
    fr[0] = (_Float16)pi[0];
    fr[1] = (_Float16)pi[1];
    fr[2] = (_Float16)(pj[0] - pi[0]);
    fr[3] = (_Float16)(pj[1] - pi[1]);
  }
  {  // W1t: wbuf[n][kk] = w1[kk][n], kk<4, rest stays zero
    const int n = tid >> 2, kk = tid & 3;
    wbuf[n * 64 + kk] = (_Float16)w1[kk * 64 + n];
  }
  __syncthreads();
  // layer 1: M=160 (10 tiles) x N=64 (4 tiles), K=32 (1 step)
#pragma unroll
  for (int i = 0; i < 5; ++i) {
    const int t = i * 8 + wid, mt = t >> 2, nt = t & 3;
    v8f acc = {};
    acc = wmma16(ld_frag(feat, mt * 16, 0, 32), ld_frag(wbuf, nt * 16, 0, 64), acc);
    const int col = nt * 16 + (lane & 15), mh = lane >> 4;
    const float bs = b1[col];
#pragma unroll
    for (int r = 0; r < 8; ++r)
      bufA[(mt * 16 + r + 8 * mh) * 64 + col] = (_Float16)fmaxf(acc[r] + bs, 0.f);
  }
  __syncthreads();
  for (int t = tid; t < 64 * 64; t += 256)
    wbuf[t] = (_Float16)w2[(t & 63) * 64 + (t >> 6)];
  __syncthreads();
  // layer 2: K=64 (2 steps)
#pragma unroll
  for (int i = 0; i < 5; ++i) {
    const int t = i * 8 + wid, mt = t >> 2, nt = t & 3;
    v8f acc = {};
#pragma unroll
    for (int ks = 0; ks < 2; ++ks)
      acc = wmma16(ld_frag(bufA, mt * 16, ks * 32, 64),
                   ld_frag(wbuf, nt * 16, ks * 32, 64), acc);
    const int col = nt * 16 + (lane & 15), mh = lane >> 4;
    const float bs = b2[col];
#pragma unroll
    for (int r = 0; r < 8; ++r)
      bufB[(mt * 16 + r + 8 * mh) * 64 + col] = (_Float16)fmaxf(acc[r] + bs, 0.f);
  }
  __syncthreads();
  for (int t = tid; t < 64 * 64; t += 256)
    wbuf[t] = (_Float16)w3[(t & 63) * 64 + (t >> 6)];
  __syncthreads();
  // layer 3: no relu, write back into bufA
#pragma unroll
  for (int i = 0; i < 5; ++i) {
    const int t = i * 8 + wid, mt = t >> 2, nt = t & 3;
    v8f acc = {};
#pragma unroll
    for (int ks = 0; ks < 2; ++ks)
      acc = wmma16(ld_frag(bufB, mt * 16, ks * 32, 64),
                   ld_frag(wbuf, nt * 16, ks * 32, 64), acc);
    const int col = nt * 16 + (lane & 15), mh = lane >> 4;
    const float bs = b3[col];
#pragma unroll
    for (int r = 0; r < 8; ++r)
      bufA[(mt * 16 + r + 8 * mh) * 64 + col] = (_Float16)(acc[r] + bs);
  }
  __syncthreads();
  // max over the 20 neighbors of each point
  for (int t = tid; t < 8 * 64; t += 256) {
    const int p = t >> 6, c = t & 63, n = pbase + p;
    float m = -3.4e38f;
    for (int k = 0; k < KNN; ++k)
      m = fmaxf(m, (float)bufA[(p * KNN + k) * 64 + c]);
    x1f[((size_t)b * NN + n) * 64 + c] = m;
    x1h[((size_t)b * NN + n) * 64 + c] = (_Float16)m;
  }
}

// ------------------------------------------------------------- row norms of x1
__global__ __launch_bounds__(256) void d2_kernel(const float* __restrict__ x1f,
                                                 float* __restrict__ d2) {
  const int i = blockIdx.x * blockDim.x + threadIdx.x;
  if (i < BB * NN) {
    const float* p = x1f + (size_t)i * 64;
    float s = 0.f;
    for (int c = 0; c < 64; ++c) s += p[c] * p[c];
    d2[i] = s;
  }
}

// ------------------------------------------------- KNN #2 (64-D, WMMA gram)
__global__ __launch_bounds__(256) void knn2_kernel(
    const _Float16* __restrict__ x1h, const float* __restrict__ d2,
    int* __restrict__ idx2) {
  __shared__ alignas(16) _Float16 qa[64 * 64];
  __shared__ alignas(16) _Float16 cb[64 * 64];
  __shared__ float distb[64 * 64];
  __shared__ float d2i[64];
  __shared__ float d2j[64];
  const int tid = threadIdx.x, wid = tid >> 5, lane = tid & 31;
  const int b = blockIdx.x, i0 = blockIdx.y * 64;
  const _Float16* xb = x1h + (size_t)b * NN * 64;
  const float* db = d2 + (size_t)b * NN;
  for (int t = tid; t < 64 * 64; t += 256) qa[t] = xb[(size_t)i0 * 64 + t];
  if (tid < 64) d2i[tid] = db[i0 + tid];
  float bd[KNN]; int bi[KNN];
#pragma unroll
  for (int k = 0; k < KNN; ++k) { bd[k] = 3.4e38f; bi[k] = 0; }
  for (int jb = 0; jb < 64; ++jb) {
    __syncthreads();   // protects qa first time, cb/distb on later iterations
    for (int t = tid; t < 64 * 64; t += 256) cb[t] = xb[(size_t)jb * 64 * 64 + t];
    if (tid < 64) d2j[tid] = db[jb * 64 + tid];
    __syncthreads();
#pragma unroll
    for (int i = 0; i < 2; ++i) {
      const int t = i * 8 + wid, mt = t >> 2, nt = t & 3;
      v8f acc = {};
#pragma unroll
      for (int ks = 0; ks < 2; ++ks)
        acc = wmma16(ld_frag(qa, mt * 16, ks * 32, 64),
                     ld_frag(cb, nt * 16, ks * 32, 64), acc);
      const int col = nt * 16 + (lane & 15), mh = lane >> 4;
      const float dj = d2j[col];
#pragma unroll
      for (int r = 0; r < 8; ++r) {
        const int row = mt * 16 + r + 8 * mh;
        distb[row * 64 + col] = d2i[row] + dj - 2.f * acc[r];
      }
    }
    __syncthreads();
    if (tid < 64) {
      const int base = jb * 64;
      for (int c = 0; c < 64; ++c) {
        const float d = distb[tid * 64 + c];
        if (d < bd[KNN - 1]) {
          int p = KNN - 1;
          while (p > 0 && bd[p - 1] > d) { bd[p] = bd[p - 1]; bi[p] = bi[p - 1]; --p; }
          bd[p] = d; bi[p] = base + c;
        }
      }
    }
  }
  if (tid < 64) {
    int* o = idx2 + ((size_t)b * NN + i0 + tid) * KNN;
    for (int k = 0; k < KNN; ++k) o[k] = bi[k];
  }
}

// ---------------------------------------------------- EdgeConv #2 (128 -> 128)
// 4 points/WG -> 80 rows; accumulators deferred in VGPRs so feat LDS is reused.
__global__ __launch_bounds__(256) void edgeconv2_kernel(
    const float* __restrict__ x1f, const int* __restrict__ idx2,
    const float* __restrict__ w, const float* __restrict__ bias,
    _Float16* __restrict__ x2h) {
  __shared__ alignas(16) _Float16 feat[80 * 128];
  __shared__ alignas(16) _Float16 wbuf[128 * 128];
  const int tid = threadIdx.x, wid = tid >> 5, lane = tid & 31;
  const int blk = blockIdx.x;
  const int b = blk >> 10;              // 1024 blocks per batch
  const int pbase = (blk & 1023) * 4;
  for (int t = tid; t < 128 * 128; t += 256)
    wbuf[t] = (_Float16)w[(t & 127) * 128 + (t >> 7)];
  for (int t = tid; t < 80 * 128; t += 256) {
    const int r = t >> 7, c = t & 127;
    const int n = pbase + r / KNN;
    const int j = idx2[((size_t)b * NN + n) * KNN + (r % KNN)];
    float v;
    if (c < 64) v = x1f[((size_t)b * NN + n) * 64 + c];
    else        v = x1f[((size_t)b * NN + j) * 64 + (c - 64)] -
                    x1f[((size_t)b * NN + n) * 64 + (c - 64)];
    feat[t] = (_Float16)v;
  }
  __syncthreads();
  v8f acc[5];
#pragma unroll
  for (int i = 0; i < 5; ++i) {
    const int t = i * 8 + wid, mt = t >> 3, nt = t & 7;
    v8f a = {};
#pragma unroll
    for (int ks = 0; ks < 4; ++ks)
      a = wmma16(ld_frag(feat, mt * 16, ks * 32, 128),
                 ld_frag(wbuf, nt * 16, ks * 32, 128), a);
    acc[i] = a;
  }
  __syncthreads();
#pragma unroll
  for (int i = 0; i < 5; ++i) {
    const int t = i * 8 + wid, mt = t >> 3, nt = t & 7;
    const int col = nt * 16 + (lane & 15), mh = lane >> 4;
    const float bs = bias[col];
#pragma unroll
    for (int r = 0; r < 8; ++r)
      feat[(mt * 16 + r + 8 * mh) * 128 + col] = (_Float16)(acc[i][r] + bs);
  }
  __syncthreads();
  for (int t = tid; t < 4 * 128; t += 256) {
    const int p = t >> 7, c = t & 127;
    float m = -3.4e38f;
    for (int k = 0; k < KNN; ++k)
      m = fmaxf(m, (float)feat[(p * KNN + k) * 128 + c]);
    x2h[((size_t)b * NN + pbase + p) * 128 + c] = (_Float16)m;
  }
}

// ------------------------------------- lin1 (192 -> 1024) + partial max over N
__global__ __launch_bounds__(256) void lin1_kernel(
    const _Float16* __restrict__ x1h, const _Float16* __restrict__ x2h,
    const float* __restrict__ w, float* __restrict__ pmax) {
  __shared__ alignas(16) _Float16 abuf[64 * 192];
  __shared__ alignas(16) _Float16 wbuf[32 * 192];
  __shared__ float hbuf[64 * 32];
  const int tid = threadIdx.x, wid = tid >> 5, lane = tid & 31;
  const int b = blockIdx.x, rb = blockIdx.y;
  const int n0 = rb * 64;
  for (int t = tid; t < 64 * 192; t += 256) {
    const int r = t / 192, c = t % 192, n = n0 + r;
    abuf[t] = (c < 64) ? x1h[((size_t)b * NN + n) * 64 + c]
                       : x2h[((size_t)b * NN + n) * 128 + (c - 64)];
  }
  for (int nc = 0; nc < 32; ++nc) {
    __syncthreads();  // abuf ready (nc==0); hbuf/wbuf consumed (nc>0)
    for (int t = tid; t < 32 * 192; t += 256) {
      const int cc = t / 192, kk = t % 192;
      wbuf[t] = (_Float16)w[(size_t)kk * 1024 + nc * 32 + cc];
    }
    __syncthreads();
    {
      const int mt = wid >> 1, nt = wid & 1;
      v8f acc = {};
#pragma unroll
      for (int ks = 0; ks < 6; ++ks)
        acc = wmma16(ld_frag(abuf, mt * 16, ks * 32, 192),
                     ld_frag(wbuf, nt * 16, ks * 32, 192), acc);
      const int col = nt * 16 + (lane & 15), mh = lane >> 4;
#pragma unroll
      for (int r = 0; r < 8; ++r)
        hbuf[(mt * 16 + r + 8 * mh) * 32 + col] = acc[r];
    }
    __syncthreads();
    if (tid < 32) {
      float m = -3.4e38f;
      for (int r = 0; r < 64; ++r) m = fmaxf(m, hbuf[r * 32 + tid]);
      pmax[((size_t)b * 64 + rb) * 1024 + nc * 32 + tid] = m;
    }
  }
}

// -------------------------------- final reduce + bias + MLP head + log_softmax
__global__ __launch_bounds__(256) void head_kernel(
    const float* __restrict__ pmax, const float* __restrict__ lb,
    const float* __restrict__ mw1, const float* __restrict__ mb1,
    const float* __restrict__ mw2, const float* __restrict__ mb2,
    const float* __restrict__ mw3, const float* __restrict__ mb3,
    float* __restrict__ out) {
  __shared__ float pooled[1024];
  __shared__ float h1[512];
  __shared__ float h2[256];
  __shared__ float h3[10];
  __shared__ float red[2];
  const int b = blockIdx.x, tid = threadIdx.x;
  for (int c = tid; c < 1024; c += 256) {
    float m = -3.4e38f;
    const float* p = pmax + (size_t)b * 64 * 1024 + c;
    for (int rb = 0; rb < 64; ++rb) m = fmaxf(m, p[rb * 1024]);
    pooled[c] = m + lb[c];     // bias commutes with max over rows
  }
  __syncthreads();
  for (int c = tid; c < 512; c += 256) {
    float s = mb1[c];
    for (int k = 0; k < 1024; ++k) s += pooled[k] * mw1[(size_t)k * 512 + c];
    h1[c] = fmaxf(s, 0.f);
  }
  __syncthreads();
  for (int c = tid; c < 256; c += 256) {
    float s = mb2[c];
    for (int k = 0; k < 512; ++k) s += h1[k] * mw2[(size_t)k * 256 + c];
    h2[c] = fmaxf(s, 0.f);
  }
  __syncthreads();
  if (tid < 10) {
    float s = mb3[tid];
    for (int k = 0; k < 256; ++k) s += h2[k] * mw3[(size_t)k * 10 + tid];
    h3[tid] = s;
  }
  __syncthreads();
  if (tid == 0) {
    float m = h3[0];
    for (int k = 1; k < 10; ++k) m = fmaxf(m, h3[k]);
    float s = 0.f;
    for (int k = 0; k < 10; ++k) s += expf(h3[k] - m);
    red[0] = m; red[1] = logf(s);
  }
  __syncthreads();
  if (tid < 10) out[b * 10 + tid] = h3[tid] - red[0] - red[1];
}

extern "C" void kernel_launch(void* const* d_in, const int* in_sizes, int n_in,
                              void* d_out, int out_size, void* d_ws, size_t ws_size,
                              hipStream_t stream) {
  (void)in_sizes; (void)n_in; (void)out_size; (void)ws_size;
  const float* data  = (const float*)d_in[0];
  const float* c1w1  = (const float*)d_in[1];
  const float* c1b1  = (const float*)d_in[2];
  const float* c1w2  = (const float*)d_in[3];
  const float* c1b2  = (const float*)d_in[4];
  const float* c1w3  = (const float*)d_in[5];
  const float* c1b3  = (const float*)d_in[6];
  const float* c2w1  = (const float*)d_in[7];
  const float* c2b1  = (const float*)d_in[8];
  const float* lin1w = (const float*)d_in[9];
  const float* lin1b = (const float*)d_in[10];
  const float* mw1   = (const float*)d_in[11];
  const float* mb1   = (const float*)d_in[12];
  const float* mw2   = (const float*)d_in[13];
  const float* mb2   = (const float*)d_in[14];
  const float* mw3   = (const float*)d_in[15];
  const float* mb3   = (const float*)d_in[16];

  char* ws = (char*)d_ws;
  size_t off = 0;
  auto alloc = [&](size_t bytes) -> void* {
    void* p = ws + off;
    off = (off + bytes + 255) & ~(size_t)255;
    return p;
  };
  int*      idx1 = (int*)alloc((size_t)BB * NN * KNN * 4);
  float*    x1f  = (float*)alloc((size_t)BB * NN * 64 * 4);
  _Float16* x1h  = (_Float16*)alloc((size_t)BB * NN * 64 * 2);
  float*    d2   = (float*)alloc((size_t)BB * NN * 4);
  int*      idx2 = (int*)alloc((size_t)BB * NN * KNN * 4);
  _Float16* x2h  = (_Float16*)alloc((size_t)BB * NN * 128 * 2);
  float*    pm   = (float*)alloc((size_t)BB * 64 * 1024 * 4);

  knn1_kernel<<<dim3(BB, NN / 128), 128, 0, stream>>>(data, idx1);
  edgeconv1_kernel<<<dim3(BB * NN / 8), 256, 0, stream>>>(
      data, idx1, c1w1, c1b1, c1w2, c1b2, c1w3, c1b3, x1f, x1h);
  d2_kernel<<<dim3(BB * NN / 256), 256, 0, stream>>>(x1f, d2);
  knn2_kernel<<<dim3(BB, NN / 64), 256, 0, stream>>>(x1h, d2, idx2);
  edgeconv2_kernel<<<dim3(BB * NN / 4), 256, 0, stream>>>(x1f, idx2, c2w1, c2b1, x2h);
  lin1_kernel<<<dim3(BB, 64), 256, 0, stream>>>(x1h, x2h, lin1w, pm);
  head_kernel<<<dim3(BB), 256, 0, stream>>>(pm, lin1b, mw1, mb1, mw2, mb2, mw3,
                                            mb3, (float*)d_out);
}
